// ConvAttention_29721173688648
// MI455X (gfx1250) — matrix operand order repaired and checked by reading, not compiled, not run on hardware
//
#include <hip/hip_runtime.h>
#include <math.h>

typedef __bf16 bf16;
typedef bf16  v16bf __attribute__((ext_vector_type(16)));
typedef bf16  v8bf  __attribute__((ext_vector_type(8)));
typedef float v8f   __attribute__((ext_vector_type(8)));

#define B_    8
#define NTOK  1026     // tokens inside attention (1025 + 1 pad)
#define NDROP 1025     // tokens after dropping the pad
#define TPAD  1056     // token dim padded to 32 for guard-free loads
#define C_    384
#define H_    6
#define QKVC  1152
#define HID_  1536
#define WWIN  256
#define POOLK 3456
#define M3    257
#define SCALE 0.40824829046386307f  /* 6^-0.5 */

// ---------------------------------------------------------------- WMMA core
static __device__ __forceinline__ v8f wmma_bf16(v16bf a, v16bf b, v8f c) {
  return __builtin_amdgcn_wmma_f32_16x16x32_bf16(false, a, false, b, (short)0, c,
                                                 false, false);
}

// A-frag (16x32 bf16): per lane two contiguous 16B chunks of its (clamped) row.
// rowp = &A[row * stride]; chunks at k0+lo and k0+lo+16, lo = 8*(lane>=16).
static __device__ __forceinline__ v16bf load_a16(const bf16* rowp, int k0) {
  int lo = (threadIdx.x & 16) ? 8 : 0;
  v8bf a0 = *(const v8bf*)(rowp + k0 + lo);
  v8bf a1 = *(const v8bf*)(rowp + k0 + lo + 16);
  return __builtin_shufflevector(a0, a1, 0, 1, 2, 3, 4, 5, 6, 7,
                                 8, 9, 10, 11, 12, 13, 14, 15);
}

// B-frag (32x16 bf16): per lane two contiguous 16B chunks of its column's
// K-major data. colp = &B[n * strideK]; chunks at k0+lo and k0+lo+16, lo = 8*(lane>=16).
static __device__ __forceinline__ v16bf load_b16(const bf16* colp, int k0) {
  int lo = (threadIdx.x & 16) ? 8 : 0;
  v8bf b0 = *(const v8bf*)(colp + k0 + lo);
  v8bf b1 = *(const v8bf*)(colp + k0 + lo + 16);
  return __builtin_shufflevector(b0, b1, 0, 1, 2, 3, 4, 5, 6, 7,
                                 8, 9, 10, 11, 12, 13, 14, 15);
}

static __device__ __forceinline__ float shfl_max16(float v) {
  v = fmaxf(v, __shfl_xor(v, 1));
  v = fmaxf(v, __shfl_xor(v, 2));
  v = fmaxf(v, __shfl_xor(v, 4));
  v = fmaxf(v, __shfl_xor(v, 8));
  return v;
}
static __device__ __forceinline__ float shfl_sum16(float v) {
  v += __shfl_xor(v, 1);
  v += __shfl_xor(v, 2);
  v += __shfl_xor(v, 4);
  v += __shfl_xor(v, 8);
  return v;
}

static __device__ __forceinline__ float gelu_exact(float x) {
  return 0.5f * x * (1.0f + erff(x * 0.70710678118654752f));
}
static __device__ __forceinline__ int imin(int a, int b) { return a < b ? a : b; }

// ------------------------------------------------------- mask / index setup
// mask codes: 1 -> keep scaled score, 0 -> force score 0.0, -1 -> -inf
__global__ void mask_init_kernel(signed char* maskc) {
  for (long i = blockIdx.x * 256 + threadIdx.x; i < (long)NTOK * NTOK;
       i += (long)gridDim.x * 256) {
    int row = (int)(i / NTOK), col = (int)(i % NTOK);
    signed char cd;
    if (row == NTOK - 1 || col == NTOK - 1) cd = 0;
    else if (row == 0 || col == 0)          cd = 1;
    else                                    cd = -1;
    maskc[i] = cd;
  }
}

__global__ void mask_fill_kernel(signed char* maskc) {
  int w = blockIdx.x;             // 256 windows
  int t = threadIdx.x;            // 128 threads, 81 pairs
  if (t >= 81) return;
  int wi = w >> 4, wj = w & 15;
  int ja = t / 9, jb = t % 9;
  int ra = 2 * wi + ja / 3 - 1, ca = 2 * wj + ja % 3 - 1;
  int rb = 2 * wi + jb / 3 - 1, cb = 2 * wj + jb % 3 - 1;
  bool va = (ra >= 0 && ra < 32 && ca >= 0 && ca < 32);
  bool vb = (rb >= 0 && rb < 32 && cb >= 0 && cb < 32);
  if (va && vb) {
    int ta = ra * 32 + ca, tb = rb * 32 + cb;
    maskc[(long)(ta + 1) * NTOK + (tb + 1)] = 1;
  }
}

__global__ void build_idx_kernel(int* gidx) {
  int i = blockIdx.x * 256 + threadIdx.x;
  if (i >= WWIN * 9) return;
  int w = i / 9, j = i % 9;
  int wi = w >> 4, wj = w & 15;
  int r = 2 * wi + j / 3 - 1, c = 2 * wj + j % 3 - 1;
  gidx[i] = (r >= 0 && r < 32 && c >= 0 && c < 32) ? (r * 32 + c + 1) : 1024;
}

// ------------------------------------------------------------ weight prep
// straight f32 -> bf16 cast, layout preserved (N,K) row-major == K-contiguous
__global__ void convert_bf16_kernel(const float* W, bf16* Wb, long n) {
  for (long i = blockIdx.x * 256 + threadIdx.x; i < n; i += (long)gridDim.x * 256)
    Wb[i] = (bf16)W[i];
}

// V transpose: vt[(b*H+h)*64 + d][t] over TPAD tokens, zero padded
__global__ void vtrans_kernel(const bf16* qkvb, bf16* vt) {
  for (long i = blockIdx.x * 256 + threadIdx.x; i < (long)B_ * H_ * 64 * TPAD;
       i += (long)gridDim.x * 256) {
    int t  = (int)(i % TPAD);
    long r = i / TPAD;          // (b*H+h)*64 + d
    int d  = (int)(r & 63);
    long bh = r >> 6;
    int hh = (int)(bh % H_);
    int b  = (int)(bh / H_);
    bf16 val = (bf16)0.0f;
    if (t < NTOK)
      val = qkvb[((long)b * NTOK + t) * QKVC + 2 * C_ + hh * 64 + d];
    vt[i] = val;
  }
}

// ------------------------------------------------------------ layernorms
__global__ __launch_bounds__(128) void ln1_pad_kernel(const float* x, const float* w,
                                                      const float* bb, float* h,
                                                      bf16* hb) {
  int r = blockIdx.x, b = blockIdx.y, tid = threadIdx.x;
  long orow = ((long)b * NTOK + r) * C_;
  if (r == NTOK - 1) {
    for (int c = tid; c < C_; c += 128) { h[orow + c] = 0.f; hb[orow + c] = (bf16)0.0f; }
    return;
  }
  long irow = ((long)b * NDROP + r) * C_;
  __shared__ float red[128];
  float v0 = x[irow + tid], v1 = x[irow + tid + 128], v2 = x[irow + tid + 256];
  red[tid] = v0 + v1 + v2;
  __syncthreads();
  for (int s = 64; s > 0; s >>= 1) { if (tid < s) red[tid] += red[tid + s]; __syncthreads(); }
  float mu = red[0] / (float)C_;
  __syncthreads();
  float d0 = v0 - mu, d1 = v1 - mu, d2 = v2 - mu;
  red[tid] = d0 * d0 + d1 * d1 + d2 * d2;
  __syncthreads();
  for (int s = 64; s > 0; s >>= 1) { if (tid < s) red[tid] += red[tid + s]; __syncthreads(); }
  float rstd = rsqrtf(red[0] / (float)C_ + 1e-5f);
  float o0 = d0 * rstd * w[tid] + bb[tid];
  float o1 = d1 * rstd * w[tid + 128] + bb[tid + 128];
  float o2 = d2 * rstd * w[tid + 256] + bb[tid + 256];
  h[orow + tid] = o0;        hb[orow + tid]       = (bf16)o0;
  h[orow + tid + 128] = o1;  hb[orow + tid + 128] = (bf16)o1;
  h[orow + tid + 256] = o2;  hb[orow + tid + 256] = (bf16)o2;
}

__global__ __launch_bounds__(128) void ln2_kernel(const float* h2, const float* w,
                                                  const float* bb, float* h2n,
                                                  bf16* h2nb) {
  int r = blockIdx.x, b = blockIdx.y, tid = threadIdx.x;
  long irow = ((long)b * NTOK + r) * C_;
  long orow = ((long)b * NDROP + r) * C_;
  __shared__ float red[128];
  float v0 = h2[irow + tid], v1 = h2[irow + tid + 128], v2 = h2[irow + tid + 256];
  red[tid] = v0 + v1 + v2;
  __syncthreads();
  for (int s = 64; s > 0; s >>= 1) { if (tid < s) red[tid] += red[tid + s]; __syncthreads(); }
  float mu = red[0] / (float)C_;
  __syncthreads();
  float d0 = v0 - mu, d1 = v1 - mu, d2 = v2 - mu;
  red[tid] = d0 * d0 + d1 * d1 + d2 * d2;
  __syncthreads();
  for (int s = 64; s > 0; s >>= 1) { if (tid < s) red[tid] += red[tid + s]; __syncthreads(); }
  float rstd = rsqrtf(red[0] / (float)C_ + 1e-5f);
  float o0 = d0 * rstd * w[tid] + bb[tid];
  float o1 = d1 * rstd * w[tid + 128] + bb[tid + 128];
  float o2 = d2 * rstd * w[tid + 256] + bb[tid + 256];
  h2n[orow + tid] = o0;        h2nb[orow + tid]       = (bf16)o0;
  h2n[orow + tid + 128] = o1;  h2nb[orow + tid + 128] = (bf16)o1;
  h2n[orow + tid + 256] = o2;  h2nb[orow + tid + 256] = (bf16)o2;
}

__global__ __launch_bounds__(128) void ln3_kernel(const float* h2n, const float* tok,
                                                  const float* w, const float* bb,
                                                  bf16* h3nb) {
  int r = blockIdx.x, b = blockIdx.y, tid = threadIdx.x;
  const float* src = (r == 0) ? (h2n + (long)b * NDROP * C_)
                              : (tok + ((long)b * WWIN + (r - 1)) * C_);
  long orow = ((long)b * M3 + r) * C_;
  __shared__ float red[128];
  float v0 = src[tid], v1 = src[tid + 128], v2 = src[tid + 256];
  red[tid] = v0 + v1 + v2;
  __syncthreads();
  for (int s = 64; s > 0; s >>= 1) { if (tid < s) red[tid] += red[tid + s]; __syncthreads(); }
  float mu = red[0] / (float)C_;
  __syncthreads();
  float d0 = v0 - mu, d1 = v1 - mu, d2 = v2 - mu;
  red[tid] = d0 * d0 + d1 * d1 + d2 * d2;
  __syncthreads();
  for (int s = 64; s > 0; s >>= 1) { if (tid < s) red[tid] += red[tid + s]; __syncthreads(); }
  float rstd = rsqrtf(red[0] / (float)C_ + 1e-5f);
  h3nb[orow + tid]       = (bf16)(d0 * rstd * w[tid] + bb[tid]);
  h3nb[orow + tid + 128] = (bf16)(d1 * rstd * w[tid + 128] + bb[tid + 128]);
  h3nb[orow + tid + 256] = (bf16)(d2 * rstd * w[tid + 256] + bb[tid + 256]);
}

// ---------------------------------------------------- generic WMMA GEMM
// One wave computes a 16x64 output tile (one A-frag feeds 4 WMMAs per k-step).
// Wb is bf16 row-major (N,K)  (K-contiguous per output column).
// K must be a multiple of 32, N a multiple of 64.
__global__ __launch_bounds__(32)
void gemm_wmma_kernel(const bf16* A, long long bsA, int lda,
                      const bf16* Bw,
                      const float* bias,
                      const float* resid, long long bsR,
                      float* C, long long bsC,
                      bf16* Cb, long long bsCb,
                      int M, int N, int Kd, int act) {
  int lane = threadIdx.x;
  int cl = lane & 15;
  int n0 = blockIdx.x * 64, m0 = blockIdx.y * 16, bz = blockIdx.z;
  int mrow = imin(m0 + cl, M - 1);                 // clamp: tail rows masked at store
  const bf16* ap  = A + (long)bz * bsA + (long)mrow * lda;
  const bf16* bp0 = Bw + (long)(n0 + cl) * Kd;
  const bf16* bp1 = bp0 + (long)16 * Kd;
  const bf16* bp2 = bp0 + (long)32 * Kd;
  const bf16* bp3 = bp0 + (long)48 * Kd;
  v8f c0 = {}, c1 = {}, c2 = {}, c3 = {};
  for (int k0 = 0; k0 < Kd; k0 += 32) {
    v16bf a = load_a16(ap, k0);
    c0 = wmma_bf16(a, load_b16(bp0, k0), c0);
    c1 = wmma_bf16(a, load_b16(bp1, k0), c1);
    c2 = wmma_bf16(a, load_b16(bp2, k0), c2);
    c3 = wmma_bf16(a, load_b16(bp3, k0), c3);
  }
  int rb = (lane & 16) ? 8 : 0;
#pragma unroll
  for (int v = 0; v < 8; ++v) {
    int row = m0 + v + rb;
    if (row < M) {
      float vals[4] = {c0[v], c1[v], c2[v], c3[v]};
#pragma unroll
      for (int j = 0; j < 4; ++j) {
        int nn = n0 + 16 * j + cl;
        float val = vals[j];
        if (bias)  val += bias[nn];
        if (resid) val += resid[(long)bz * bsR + (long)row * N + nn];
        if (act)   val = gelu_exact(val);
        if (C)  C[(long)bz * bsC + (long)row * N + nn] = val;
        if (Cb) Cb[(long)bz * bsCb + (long)row * N + nn] = (bf16)val;
      }
    }
  }
}

// ------------------------------------- pool GEMM with fused window gather
// A-frag chunks (8 contiguous k) never cross a 384 boundary (8 | 384).
static __device__ __forceinline__ v16bf
load_a_gather(const bf16* H, const int* gw, int k0) {
  int lo = (threadIdx.x & 16) ? 8 : 0;
  int ka = k0 + lo, kb = k0 + lo + 16;
  int ja = ka / C_, jb = kb / C_;
  const bf16* pa = H + (long)gw[ja] * C_ + (ka - ja * C_);
  const bf16* pb = H + (long)gw[jb] * C_ + (kb - jb * C_);
  v8bf a0 = *(const v8bf*)pa;
  v8bf a1 = *(const v8bf*)pb;
  return __builtin_shufflevector(a0, a1, 0, 1, 2, 3, 4, 5, 6, 7,
                                 8, 9, 10, 11, 12, 13, 14, 15);
}

__global__ __launch_bounds__(32)
void pool_gemm_kernel(const bf16* H2nb, const int* gidx, const bf16* Wb,
                      const float* bias, float* tok) {
  int lane = threadIdx.x;
  int cl = lane & 15;
  int n0 = blockIdx.x * 64, m0 = blockIdx.y * 16, b = blockIdx.z;
  const bf16* H = H2nb + (long)b * NDROP * C_;
  const int* gw = gidx + (m0 + cl) * 9;     // this lane's window indices
  const bf16* bp0 = Wb + (long)(n0 + cl) * POOLK;
  const bf16* bp1 = bp0 + (long)16 * POOLK;
  const bf16* bp2 = bp0 + (long)32 * POOLK;
  const bf16* bp3 = bp0 + (long)48 * POOLK;
  v8f c0 = {}, c1 = {}, c2 = {}, c3 = {};
  for (int k0 = 0; k0 < POOLK; k0 += 32) {
    v16bf a = load_a_gather(H, gw, k0);
    c0 = wmma_bf16(a, load_b16(bp0, k0), c0);
    c1 = wmma_bf16(a, load_b16(bp1, k0), c1);
    c2 = wmma_bf16(a, load_b16(bp2, k0), c2);
    c3 = wmma_bf16(a, load_b16(bp3, k0), c3);
  }
  int rb = (lane & 16) ? 8 : 0;
#pragma unroll
  for (int v = 0; v < 8; ++v) {
    int row = m0 + v + rb;
    float vals[4] = {c0[v], c1[v], c2[v], c3[v]};
#pragma unroll
    for (int j = 0; j < 4; ++j) {
      int nn = n0 + 16 * j + cl;
      tok[((long)b * WWIN + row) * C_ + nn] = vals[j] + bias[nn];
    }
  }
}

// ---------------------------------------- flash attention (one wave/q-tile)
__global__ __launch_bounds__(32)
void attn_kernel(const bf16* qkvb, const bf16* vt, const signed char* maskc,
                 bf16* attnb) {
  int lane = threadIdx.x;
  int cl = lane & 15;
  int rb = (lane & 16) ? 8 : 0;
  int q0 = blockIdx.x * 16;      // 65 tiles cover 1026 rows
  int hh = blockIdx.y;           // head
  int b  = blockIdx.z;
  const bf16* base = qkvb + (long)b * NTOK * QKVC;
  const bf16* qp = base + hh * 64;
  const bf16* kp = base + C_ + hh * 64;
  const bf16* vtp = vt + ((long)(b * H_ + hh) * 64) * TPAD;  // [d][t]
  __shared__ bf16 s_p[16 * 32];

  const bf16* aqp = qp + (long)imin(q0 + cl, NTOK - 1) * QKVC;
  v16bf aq0 = load_a16(aqp, 0);
  v16bf aq1 = load_a16(aqp, 32);

  // per-lane PV B-frag column pointers (d = dt*16 + cl), token-contiguous
  const bf16* vp0 = vtp + (long)(0  + cl) * TPAD;
  const bf16* vp1 = vtp + (long)(16 + cl) * TPAD;
  const bf16* vp2 = vtp + (long)(32 + cl) * TPAD;
  const bf16* vp3 = vtp + (long)(48 + cl) * TPAD;

  v8f acc0 = {}, acc1 = {}, acc2 = {}, acc3 = {};
  float m_r[8], l_r[8];
#pragma unroll
  for (int v = 0; v < 8; ++v) { m_r[v] = -INFINITY; l_r[v] = 0.f; }

  for (int t0 = 0; t0 < TPAD; t0 += 32) {   // 33 steps cover 1026 (+pad)
    // --- scores for 32 key tokens: two 16-col tiles, K-dim 64
    const bf16* kc0 = kp + (long)imin(t0 + cl, NTOK - 1) * QKVC;       // clamped
    const bf16* kc1 = kp + (long)imin(t0 + 16 + cl, NTOK - 1) * QKVC;
    v8f s0 = {}, s1 = {};
    s0 = wmma_bf16(aq0, load_b16(kc0, 0), s0);
    s0 = wmma_bf16(aq1, load_b16(kc0, 32), s0);
    s1 = wmma_bf16(aq0, load_b16(kc1, 0), s1);
    s1 = wmma_bf16(aq1, load_b16(kc1, 32), s1);

    // --- mask, online softmax update, stage P into LDS
    int c0 = t0 + cl, c1 = t0 + 16 + cl;
#pragma unroll
    for (int v = 0; v < 8; ++v) {
      int row = q0 + v + rb;
      const signed char* mrow = maskc + (long)imin(row, NTOK - 1) * NTOK;
      float sv0, sv1;
      if (c0 < NTOK) {
        signed char cd = mrow[c0];
        sv0 = (cd == 1) ? s0[v] * SCALE : (cd == 0 ? 0.f : -INFINITY);
      } else sv0 = -INFINITY;
      if (c1 < NTOK) {
        signed char cd = mrow[c1];
        sv1 = (cd == 1) ? s1[v] * SCALE : (cd == 0 ? 0.f : -INFINITY);
      } else sv1 = -INFINITY;

      float mx    = shfl_max16(fmaxf(sv0, sv1));
      float m_new = fmaxf(m_r[v], mx);
      float alpha = (m_new == -INFINITY) ? 0.f : __expf(m_r[v] - m_new);
      float p0 = (m_new == -INFINITY) ? 0.f : __expf(sv0 - m_new);
      float p1 = (m_new == -INFINITY) ? 0.f : __expf(sv1 - m_new);
      float rs = shfl_sum16(p0 + p1);
      l_r[v] = l_r[v] * alpha + rs;
      m_r[v] = m_new;
      acc0[v] *= alpha; acc1[v] *= alpha; acc2[v] *= alpha; acc3[v] *= alpha;
      s_p[(v + rb) * 32 + cl]      = (bf16)p0;
      s_p[(v + rb) * 32 + cl + 16] = (bf16)p1;
    }
    __syncthreads();

    // --- P (16x32) @ V (32x64): four 16-col d-tiles, guard-free vT loads
    v16bf ap = load_a16(s_p + cl * 32, 0);
    acc0 = wmma_bf16(ap, load_b16(vp0, t0), acc0);
    acc1 = wmma_bf16(ap, load_b16(vp1, t0), acc1);
    acc2 = wmma_bf16(ap, load_b16(vp2, t0), acc2);
    acc3 = wmma_bf16(ap, load_b16(vp3, t0), acc3);
    __syncthreads();
  }

#pragma unroll
  for (int v = 0; v < 8; ++v) {
    int row = q0 + v + rb;
    if (row < NTOK) {
      float inv = 1.f / l_r[v];
      long o = ((long)b * NTOK + row) * C_ + hh * 64;
      attnb[o + cl]      = (bf16)(acc0[v] * inv);
      attnb[o + 16 + cl] = (bf16)(acc1[v] * inv);
      attnb[o + 32 + cl] = (bf16)(acc2[v] * inv);
      attnb[o + 48 + cl] = (bf16)(acc3[v] * inv);
    }
  }
}

// ------------------------------------------------------------ orchestration
extern "C" void kernel_launch(void* const* d_in, const int* in_sizes, int n_in,
                              void* d_out, int out_size, void* d_ws, size_t ws_size,
                              hipStream_t stream) {
  const float* x       = (const float*)d_in[0];
  const float* norm1_w = (const float*)d_in[1];
  const float* norm1_b = (const float*)d_in[2];
  const float* qkv_w   = (const float*)d_in[3];
  const float* proj_w  = (const float*)d_in[4];
  const float* proj_b  = (const float*)d_in[5];
  const float* norm2_w = (const float*)d_in[6];
  const float* norm2_b = (const float*)d_in[7];
  const float* pool_w  = (const float*)d_in[8];
  const float* pool_b  = (const float*)d_in[9];
  const float* norm3_w = (const float*)d_in[10];
  const float* norm3_b = (const float*)d_in[11];
  const float* fc1_w   = (const float*)d_in[12];
  const float* fc1_b   = (const float*)d_in[13];
  const float* fc2_w   = (const float*)d_in[14];
  const float* fc2_b   = (const float*)d_in[15];
  float* out = (float*)d_out;

  // workspace partition (256B aligned slabs)
  char* p = (char*)d_ws;
  auto alloc = [&](size_t bytes) -> void* {
    void* r = (void*)p;
    p += (bytes + 255) & ~(size_t)255;
    return r;
  };
  signed char* maskc = (signed char*)alloc((size_t)NTOK * NTOK);
  int*   gidx   = (int*)  alloc((size_t)WWIN * 9 * 4);
  float* h      = (float*)alloc((size_t)B_ * NTOK * C_ * 4);
  bf16*  hb     = (bf16*) alloc((size_t)B_ * NTOK * C_ * 2);
  bf16*  qkv_wb = (bf16*) alloc((size_t)QKVC * C_ * 2);
  bf16*  proj_wb= (bf16*) alloc((size_t)C_ * C_ * 2);
  bf16*  pool_wb= (bf16*) alloc((size_t)C_ * POOLK * 2);
  bf16*  fc1_wb = (bf16*) alloc((size_t)HID_ * C_ * 2);
  bf16*  fc2_wb = (bf16*) alloc((size_t)C_ * HID_ * 2);
  bf16*  qkvb   = (bf16*) alloc((size_t)B_ * NTOK * QKVC * 2);
  bf16*  vt     = (bf16*) alloc((size_t)B_ * H_ * 64 * TPAD * 2);
  bf16*  attnb  = (bf16*) alloc((size_t)B_ * NTOK * C_ * 2);
  float* h2     = (float*)alloc((size_t)B_ * NTOK * C_ * 4);
  float* h2n    = (float*)alloc((size_t)B_ * NDROP * C_ * 4);
  bf16*  h2nb   = (bf16*) alloc((size_t)B_ * NDROP * C_ * 2);
  float* tok    = (float*)alloc((size_t)B_ * WWIN * C_ * 4);
  bf16*  h3nb   = (bf16*) alloc((size_t)B_ * M3 * C_ * 2);
  bf16*  a1b    = (bf16*) alloc((size_t)B_ * M3 * HID_ * 2);

  // 1) mask + gather index construction
  mask_init_kernel<<<4113, 256, 0, stream>>>(maskc);
  mask_fill_kernel<<<WWIN, 128, 0, stream>>>(maskc);
  build_idx_kernel<<<(WWIN * 9 + 255) / 256, 256, 0, stream>>>(gidx);

  // 2) weight conversion f32 -> bf16 (layout preserved: (N,K) row-major)
  convert_bf16_kernel<<<1728, 256, 0, stream>>>(qkv_w,  qkv_wb,  (long)QKVC * C_);
  convert_bf16_kernel<<<576,  256, 0, stream>>>(proj_w, proj_wb, (long)C_ * C_);
  convert_bf16_kernel<<<5184, 256, 0, stream>>>(pool_w, pool_wb, (long)C_ * POOLK);
  convert_bf16_kernel<<<2304, 256, 0, stream>>>(fc1_w,  fc1_wb,  (long)HID_ * C_);
  convert_bf16_kernel<<<2304, 256, 0, stream>>>(fc2_w,  fc2_wb,  (long)C_ * HID_);

  // 3) LN1 + pad row
  ln1_pad_kernel<<<dim3(NTOK, B_), 128, 0, stream>>>(x, norm1_w, norm1_b, h, hb);

  // 4) qkv = hb @ qkv_w^T  -> bf16 (B,1026,1152)
  gemm_wmma_kernel<<<dim3(QKVC / 64, 65, B_), 32, 0, stream>>>(
      hb, (long long)NTOK * C_, C_, qkv_wb,
      nullptr, nullptr, 0, nullptr, 0,
      qkvb, (long long)NTOK * QKVC, NTOK, QKVC, C_, 0);

  // 5) V transpose with token padding -> vt (B,H,64,TPAD)
  vtrans_kernel<<<4096, 256, 0, stream>>>(qkvb, vt);

  // 6) masked flash attention -> attnb bf16 (B,1026,384)
  attn_kernel<<<dim3(65, H_, B_), 32, 0, stream>>>(qkvb, vt, maskc, attnb);

  // 7) proj + bias + residual -> h2 f32
  gemm_wmma_kernel<<<dim3(C_ / 64, 65, B_), 32, 0, stream>>>(
      attnb, (long long)NTOK * C_, C_, proj_wb,
      proj_b, h, (long long)NTOK * C_,
      h2, (long long)NTOK * C_, nullptr, 0,
      NTOK, C_, C_, 0);

  // 8) LN2 over first 1025 tokens
  ln2_kernel<<<dim3(NDROP, B_), 128, 0, stream>>>(h2, norm2_w, norm2_b, h2n, h2nb);

  // 9) windowed gather + pool GEMM -> tok f32 (B,256,384)
  pool_gemm_kernel<<<dim3(C_ / 64, WWIN / 16, B_), 32, 0, stream>>>(
      h2nb, gidx, pool_wb, pool_b, tok);

  // 10) LN3 over concat(cls, tok) -> h3nb bf16 (B,257,384)
  ln3_kernel<<<dim3(M3, B_), 128, 0, stream>>>(h2n, tok, norm3_w, norm3_b, h3nb);

  // 11) fc1 + gelu -> a1b bf16 (B,257,1536)
  gemm_wmma_kernel<<<dim3(HID_ / 64, 17, B_), 32, 0, stream>>>(
      h3nb, (long long)M3 * C_, C_, fc1_wb,
      fc1_b, nullptr, 0, nullptr, 0,
      a1b, (long long)M3 * HID_, M3, HID_, C_, 1);

  // 12) fc2 + bias -> out f32 (B,257,384)
  gemm_wmma_kernel<<<dim3(C_ / 64, 17, B_), 32, 0, stream>>>(
      a1b, (long long)M3 * HID_, HID_, fc2_wb,
      fc2_b, nullptr, 0,
      out, (long long)M3 * C_, nullptr, 0,
      M3, C_, HID_, 0);
}
